// VarLengthMultiheadSA_31808527794808
// MI455X (gfx1250) — matrix operand order repaired and checked by reading, not compiled, not run on hardware
//
#include <hip/hip_runtime.h>
#include <hip/hip_bf16.h>

// ---------------------------------------------------------------------------
// VarLengthMultiheadSA for MI455X (gfx1250, wave32, WMMA)
//
// N=40000 points, C=256, H=8 heads, HD=32, K=32 neighbors/point, M=N*K.
// Pipeline:
//   0) weights f32 -> f16, transposed (K-major) into ws
//   1) fused QKV GEMM (rows gathered by sort_idx) -> f16 q/k/v in ws
//   2) per-point/per-head attention (dot, softmax-32, weighted V) -> f16 xs
//   3) output GEMM xs @ Wp + bp, scatter rows by sort_idx -> f32 d_out
// GEMMs use v_wmma_f32_16x16x32_f16 (f32 accumulate).
// ---------------------------------------------------------------------------

typedef __attribute__((ext_vector_type(16))) _Float16 v16h;
typedef __attribute__((ext_vector_type(8)))  _Float16 v8h;
typedef __attribute__((ext_vector_type(8)))  float    v8f;

#define NPTS        40000
#define CDIM        256
#define NHEAD       8
#define HDIM        32
#define KNBR        32
#define ROWS_PER_BLK 16
#define NBLK_GEMM   (NPTS / ROWS_PER_BLK)   // 2500 (exact)
#define LDSA_STRIDE 264                      // 256 + 8 halves pad (bank spread)
#define QKV_ELEMS   ((size_t)NPTS * CDIM)    // per-matrix half count
#define W_ELEMS     (CDIM * CDIM)            // 65536
#define SCALE_Q     0.17677669529663687f     // HD^-0.5

// ---- workspace byte offsets ----
#define WS_WT_OFF   0                                    // 4 * 65536 f16
#define WS_QKV_OFF  (4 * W_ELEMS * 2)                    // 524288
#define WS_XS_OFF   (WS_QKV_OFF + 3 * (size_t)NPTS * CDIM * 2)
// total ~82.4 MB

// ---------------------------------------------------------------------------
// A-fragment: 16x32 f16, rows in LDS (row stride in halves).
// Lane 0-15: M=lane, halves[0:16]=K[0:16)
// Lane16-31: M=lane-16, halves[0:16]=K[16:32)
// ---------------------------------------------------------------------------
static __device__ inline v16h load_a_frag(const _Float16* __restrict__ tile,
                                          int lane, int kbase, int stride) {
    const int row = lane & 15;
    const int hi  = (lane >> 4) * 16;
    const _Float16* base = tile + row * stride + kbase;
    v8h lo8 = *(const v8h*)(base + hi);
    v8h hi8 = *(const v8h*)(base + hi + 8);
    v16h a;
#pragma unroll
    for (int i = 0; i < 8; ++i) { a[i] = lo8[i]; a[8 + i] = hi8[i]; }
    return a;
}

// ---------------------------------------------------------------------------
// B-fragment: 32x16 f16 from K-major (transposed) weights Wt[n*256 + k].
// Lane 0-15: N=lane, halves[0:16] = K[0:16); Lane16-31: N=lane-16, K[16:32).
// bcol points at Wt + n_global*256 + (lane>>4)*16; kbase advances the K loop.
// ---------------------------------------------------------------------------
static __device__ inline v16h load_b_frag(const _Float16* __restrict__ bcol,
                                          int kbase) {
    const v8h* p = (const v8h*)(bcol + kbase);
    v8h lo8 = p[0];
    v8h hi8 = p[1];
    v16h b;
#pragma unroll
    for (int i = 0; i < 8; ++i) { b[i] = lo8[i]; b[8 + i] = hi8[i]; }
    return b;
}

// ---------------------------------------------------------------------------
// Kernel 0: convert 4 weight matrices f32 -> f16, K-major transpose.
// Wt[w][n*256 + k] = W[w][k*256 + n]
// ---------------------------------------------------------------------------
__global__ __launch_bounds__(256) void prep_weights_kernel(
    const float* __restrict__ Wq, const float* __restrict__ Wk,
    const float* __restrict__ Wv, const float* __restrict__ Wp,
    _Float16* __restrict__ Wt) {
    int el  = blockIdx.x * 256 + threadIdx.x;   // 0 .. 4*65536-1
    int w   = el >> 16;
    int rem = el & (W_ELEMS - 1);
    int k   = rem >> 8;
    int n   = rem & 255;
    const float* W = (w == 0) ? Wq : (w == 1) ? Wk : (w == 2) ? Wv : Wp;
    Wt[(size_t)w * W_ELEMS + n * CDIM + k] = (_Float16)W[k * CDIM + n];
}

// ---------------------------------------------------------------------------
// Kernel 1: fused QKV projection.
// Block = 16 sorted rows (gathered via sort_idx), 256 thr = 8 waves.
// 48 output tiles (3 matrices x 16 col-tiles), 6 per wave.
// ---------------------------------------------------------------------------
__global__ __launch_bounds__(256) void qkv_gemm_kernel(
    const float* __restrict__ X, const int* __restrict__ sort_idx,
    const _Float16* __restrict__ Wt,      // [3][256][256] K-major f16
    const float* __restrict__ bq, const float* __restrict__ bk,
    const float* __restrict__ bv,
    _Float16* __restrict__ qkv) {         // [3][N][256] f16 (sorted order)
    __shared__ _Float16 Atile[ROWS_PER_BLK * LDSA_STRIDE];

    const int tid = threadIdx.x;
    const int blk = blockIdx.x;

    // Stage 16 gathered X rows into LDS as f16 (each thread: 16 floats).
    {
        const int row = tid >> 4;
        const int c0  = (tid & 15) * 16;
        const int g   = sort_idx[blk * ROWS_PER_BLK + row];
        const float4* src = (const float4*)(X + (size_t)g * CDIM + c0);
        _Float16* dst = Atile + row * LDSA_STRIDE + c0;
#pragma unroll
        for (int v = 0; v < 2; ++v) {
            float4 f0 = src[v * 2 + 0];
            float4 f1 = src[v * 2 + 1];
            v8h h;
            h[0] = (_Float16)f0.x; h[1] = (_Float16)f0.y;
            h[2] = (_Float16)f0.z; h[3] = (_Float16)f0.w;
            h[4] = (_Float16)f1.x; h[5] = (_Float16)f1.y;
            h[6] = (_Float16)f1.z; h[7] = (_Float16)f1.w;
            *(v8h*)(dst + v * 8) = h;
        }
    }
    __syncthreads();

    const int wave = tid >> 5;
    const int lane = tid & 31;

    v8f acc[6];
    const v8f vzero = {0.f, 0.f, 0.f, 0.f, 0.f, 0.f, 0.f, 0.f};
    const _Float16* bcol[6];
#pragma unroll
    for (int tt = 0; tt < 6; ++tt) {
        acc[tt] = vzero;
        const int tile = wave * 6 + tt;       // 0..47
        const int m    = tile >> 4;           // matrix 0..2
        const int ct   = tile & 15;           // col tile
        bcol[tt] = Wt + (size_t)m * W_ELEMS
                      + (ct * 16 + (lane & 15)) * CDIM + (lane >> 4) * 16;
    }

#pragma unroll
    for (int ks = 0; ks < 8; ++ks) {
        const v16h a = load_a_frag(Atile, lane, ks * 32, LDSA_STRIDE);
#pragma unroll
        for (int tt = 0; tt < 6; ++tt) {
            const v16h b = load_b_frag(bcol[tt], ks * 32);
            acc[tt] = __builtin_amdgcn_wmma_f32_16x16x32_f16(
                false, a, false, b, (short)0, acc[tt], false, false);
        }
    }

    // Epilogue: bias, q-scale, f16 store in sorted order.
#pragma unroll
    for (int tt = 0; tt < 6; ++tt) {
        const int tile = wave * 6 + tt;
        const int m    = tile >> 4;
        const int ct   = tile & 15;
        const int col  = ct * 16 + (lane & 15);
        const float* bias = (m == 0) ? bq : (m == 1) ? bk : bv;
        const float bval  = bias[col];
        const float sc    = (m == 0) ? SCALE_Q : 1.0f;
        _Float16* outb = qkv + (size_t)m * QKV_ELEMS;
        const int rbase = blk * ROWS_PER_BLK + (lane >> 4) * 8;
#pragma unroll
        for (int j = 0; j < 8; ++j) {
            float vv = (acc[tt][j] + bval) * sc;
            outb[(size_t)(rbase + j) * CDIM + col] = (_Float16)vv;
        }
    }
}

// ---------------------------------------------------------------------------
// Kernel 2: attention (dot -> softmax over 32 neighbors -> weighted V).
// Block = 1 sorted point, wave = head (wave32). k/v working set (~41 MB f16)
// lives in the 192 MB L2, so the random gathers stay on-chip.
// ---------------------------------------------------------------------------
__global__ __launch_bounds__(256) void attn_kernel(
    const _Float16* __restrict__ qkv, const int* __restrict__ index_1,
    _Float16* __restrict__ xs) {
    __shared__ int   nbr[KNBR];
    __shared__ float qrow[CDIM];
    __shared__ float prob[NHEAD][KNBR];

    const int p   = blockIdx.x;
    const int tid = threadIdx.x;
    const _Float16* qS = qkv;
    const _Float16* kS = qkv + QKV_ELEMS;
    const _Float16* vS = qkv + 2 * QKV_ELEMS;

    if (tid < KNBR) nbr[tid] = index_1[(size_t)p * KNBR + tid];
    qrow[tid] = (float)qS[(size_t)p * CDIM + tid];
    __syncthreads();

    const int h    = tid >> 5;    // head
    const int lane = tid & 31;

    // Phase 1: lane = neighbor. 64B vector load of this neighbor's k row.
    const _Float16* kp = kS + (size_t)nbr[lane] * CDIM + h * HDIM;
    const v8h* kv = (const v8h*)kp;
    v8h k0 = kv[0], k1 = kv[1], k2 = kv[2], k3 = kv[3];
    const float* qh = qrow + h * HDIM;
    float dot = 0.f;
#pragma unroll
    for (int i = 0; i < 8; ++i) dot += qh[i]      * (float)k0[i];
#pragma unroll
    for (int i = 0; i < 8; ++i) dot += qh[8 + i]  * (float)k1[i];
#pragma unroll
    for (int i = 0; i < 8; ++i) dot += qh[16 + i] * (float)k2[i];
#pragma unroll
    for (int i = 0; i < 8; ++i) dot += qh[24 + i] * (float)k3[i];

    // Softmax across the 32 lanes of this wave.
    float mx = dot;
#pragma unroll
    for (int o = 16; o >= 1; o >>= 1) mx = fmaxf(mx, __shfl_xor(mx, o, 32));
    float e = __expf(dot - mx);
    float s = e;
#pragma unroll
    for (int o = 16; o >= 1; o >>= 1) s += __shfl_xor(s, o, 32);
    prob[h][lane] = e / s;
    __syncthreads();

    // Phase 2: lane = dim. Coalesced 64B v-row loads per neighbor.
    float acc = 0.f;
#pragma unroll
    for (int j = 0; j < KNBR; ++j) {
        acc += prob[h][j] * (float)vS[(size_t)nbr[j] * CDIM + h * HDIM + lane];
    }
    xs[(size_t)p * CDIM + h * HDIM + lane] = (_Float16)acc;
}

// ---------------------------------------------------------------------------
// Kernel 3: output projection xs @ Wp + bp, scatter rows by sort_idx.
// 8 waves x 2 col-tiles = 16 col tiles of 16.
// ---------------------------------------------------------------------------
__global__ __launch_bounds__(256) void proj_gemm_kernel(
    const _Float16* __restrict__ xs, const _Float16* __restrict__ WpT,
    const float* __restrict__ bp, const int* __restrict__ sort_idx,
    float* __restrict__ out) {
    __shared__ _Float16 Atile[ROWS_PER_BLK * LDSA_STRIDE];
    __shared__ int gidx[ROWS_PER_BLK];

    const int tid = threadIdx.x;
    const int blk = blockIdx.x;

    {
        const int row = tid >> 4;
        const int c0  = (tid & 15) * 16;
        const v8h* src = (const v8h*)(xs + (size_t)(blk * ROWS_PER_BLK + row) * CDIM + c0);
        v8h* dst = (v8h*)(Atile + row * LDSA_STRIDE + c0);
        dst[0] = src[0];
        dst[1] = src[1];
    }
    if (tid < ROWS_PER_BLK) gidx[tid] = sort_idx[blk * ROWS_PER_BLK + tid];
    __syncthreads();

    const int wave = tid >> 5;
    const int lane = tid & 31;

    v8f acc[2];
    const v8f vzero = {0.f, 0.f, 0.f, 0.f, 0.f, 0.f, 0.f, 0.f};
    const _Float16* bcol[2];
#pragma unroll
    for (int tt = 0; tt < 2; ++tt) {
        acc[tt] = vzero;
        const int ct = wave * 2 + tt;
        bcol[tt] = WpT + (ct * 16 + (lane & 15)) * CDIM + (lane >> 4) * 16;
    }

#pragma unroll
    for (int ks = 0; ks < 8; ++ks) {
        const v16h a = load_a_frag(Atile, lane, ks * 32, LDSA_STRIDE);
#pragma unroll
        for (int tt = 0; tt < 2; ++tt) {
            const v16h b = load_b_frag(bcol[tt], ks * 32);
            acc[tt] = __builtin_amdgcn_wmma_f32_16x16x32_f16(
                false, a, false, b, (short)0, acc[tt], false, false);
        }
    }

#pragma unroll
    for (int tt = 0; tt < 2; ++tt) {
        const int ct  = wave * 2 + tt;
        const int col = ct * 16 + (lane & 15);
        const float bval = bp[col];
        const int rtile = (lane >> 4) * 8;
#pragma unroll
        for (int j = 0; j < 8; ++j) {
            const int g = gidx[rtile + j];
            out[(size_t)g * CDIM + col] = acc[tt][j] + bval;
        }
    }
}

// ---------------------------------------------------------------------------
extern "C" void kernel_launch(void* const* d_in, const int* in_sizes, int n_in,
                              void* d_out, int out_size, void* d_ws, size_t ws_size,
                              hipStream_t stream) {
    const float* X        = (const float*)d_in[0];
    const float* Wq       = (const float*)d_in[1];
    const float* bq       = (const float*)d_in[2];
    const float* Wk       = (const float*)d_in[3];
    const float* bk       = (const float*)d_in[4];
    const float* Wv       = (const float*)d_in[5];
    const float* bv       = (const float*)d_in[6];
    const float* Wp       = (const float*)d_in[7];
    const float* bp       = (const float*)d_in[8];
    // d_in[9] = index_0 (implicit: repeat(arange(N), K)) — unused
    const int*   index_1  = (const int*)d_in[10];
    const int*   sort_idx = (const int*)d_in[11];

    char* ws = (char*)d_ws;
    _Float16* Wt  = (_Float16*)(ws + WS_WT_OFF);   // [4][256][256] K-major f16
    _Float16* qkv = (_Float16*)(ws + WS_QKV_OFF);  // [3][N][256] f16, sorted
    _Float16* xs  = (_Float16*)(ws + WS_XS_OFF);   // [N][256] f16, sorted

    prep_weights_kernel<<<(4 * W_ELEMS) / 256, 256, 0, stream>>>(Wq, Wk, Wv, Wp, Wt);
    qkv_gemm_kernel<<<NBLK_GEMM, 256, 0, stream>>>(X, sort_idx, Wt, bq, bk, bv, qkv);
    attn_kernel<<<NPTS, 256, 0, stream>>>(qkv, index_1, xs);
    proj_gemm_kernel<<<NBLK_GEMM, 256, 0, stream>>>(xs, Wt + 3 * (size_t)W_ELEMS,
                                                    bp, sort_idx, (float*)d_out);
}
